// LinearAttention_30477087932880
// MI455X (gfx1250) — matrix operand order repaired and checked by reading, not compiled, not run on hardware
//
#include <hip/hip_runtime.h>
#include <hip/hip_bf16.h>

// ---------------- problem constants ----------------
constexpr int NB    = 16;    // batch
constexpr int CIN   = 256;   // channels
constexpr int NHW   = 4096;  // 64*64 spatial
constexpr int OQKV  = 768;   // 3*hidden
constexpr int HID   = 256;   // heads*dim_head
constexpr int HEADS = 4;
constexpr int DHEAD = 64;

// ---------------- GEMM tiling ----------------
constexpr int BM = 128, BN = 128, TK = 32;
constexpr int LDK = 40;              // padded LDS row stride in ushorts (80B: 16B-aligned rows)
constexpr int NSPLIT = 16;
constexpr int NSL = NHW / NSPLIT;    // 256

typedef __attribute__((ext_vector_type(16))) __bf16 v16bf;
typedef __attribute__((ext_vector_type(8)))  float  v8f;

// ---------------- helpers ----------------
__device__ inline unsigned short f2bf1(float f) {
  __bf16 h = (__bf16)f;                       // native v_cvt (RNE)
  return __builtin_bit_cast(unsigned short, h);
}
__device__ inline unsigned int f2bf2(float lo, float hi) {
  return (unsigned int)f2bf1(lo) | ((unsigned int)f2bf1(hi) << 16);
}
__device__ inline float bf2f(unsigned int s) {
  unsigned int u = (s & 0xFFFFu) << 16;
  return __builtin_bit_cast(float, u);
}

__device__ inline v8f wmma_bf16(v16bf a, v16bf b, v8f c) {
  return __builtin_amdgcn_wmma_f32_16x16x32_bf16(false, a, false, b, (short)0, c, false, false);
}

// async global->LDS 16B copy (ASYNCcnt); lds_addr = low 32 bits of flat LDS address
__device__ inline void async_copy_b128(unsigned lds_addr, const void* gptr) {
  asm volatile("global_load_async_to_lds_b128 %0, %1, off"
               :: "v"(lds_addr), "v"((unsigned long long)(uintptr_t)gptr)
               : "memory");
}
__device__ inline void wait_async0() {
  asm volatile("s_wait_asynccnt 0x0" ::: "memory");
}

// A-fragment (16-bit A 16x32): u[0..3] = k 8g..8g+7, u[4..7] = k 16+8g..23+8g -> two b128 loads
__device__ inline v16bf frag_a(const unsigned short* row, int g) {
  union { v16bf v; uint4 q[2]; } r;
  r.q[0] = *(const uint4*)(row + (g << 3));
  r.q[1] = *(const uint4*)(row + 16 + (g << 3));
  return r.v;
}
// B-fragment (16-bit B 32x16): u[0..3] = k 8g..8g+7, u[4..7] = k 16+8g..23+8g -> two b128 loads
__device__ inline v16bf frag_b(const unsigned short* row, int g) {
  union { v16bf v; uint4 q[2]; } r;
  r.q[0] = *(const uint4*)(row + (g << 3));
  r.q[1] = *(const uint4*)(row + 16 + (g << 3));
  return r.v;
}

// ---------------- kernel 1: convert w_qkv to bf16 ----------------
__global__ void cvt_to_bf16(const float* __restrict__ w, unsigned short* __restrict__ wb, int n) {
  int i = blockIdx.x * 256 + threadIdx.x;
  if (i < n) wb[i] = f2bf1(w[i]);
}

// ---------------- kernel 2: QKV GEMM (bf16 WMMA, fp32 accum) ----------------
// C[768 x 4096] = w_qkv[768 x 256] * x_b[256 x 4096]; rows [0,256)->q, [256,512)->k, [512,768)->v
__global__ __launch_bounds__(256) void qkv_gemm(const unsigned short* __restrict__ wq,
                                                const float* __restrict__ x,
                                                unsigned short* __restrict__ qb,
                                                unsigned short* __restrict__ kb,
                                                unsigned short* __restrict__ vb) {
  __shared__ alignas(16) unsigned short As[BM * LDK];
  __shared__ alignas(16) unsigned short Bs[BN * LDK];
  const int n0 = blockIdx.x * BN;
  const int m0 = blockIdx.y * BM;
  const int b  = blockIdx.z;
  const int t  = threadIdx.x;
  const int lane = t & 31, wave = t >> 5;
  const int g = lane >> 4, ln = lane & 15;
  const int mt0 = (wave >> 1) * 2, nt0 = (wave & 1) * 4;

  v8f acc[2][4];
#pragma unroll
  for (int i = 0; i < 2; ++i)
#pragma unroll
    for (int j = 0; j < 4; ++j) acc[i][j] = (v8f){0.f,0.f,0.f,0.f,0.f,0.f,0.f,0.f};

  // per-thread staging indices
  const int ar = t >> 1, ac = (t & 1) * 16;          // A: 128 rows x 32 -> 32B per thread
  const int k2 = (t >> 4) * 2, nc = (t & 15) * 8;    // B: k-pair x 8 columns per thread
  const unsigned la = (unsigned)(uintptr_t)(As + ar * LDK + ac);

  for (int k0 = 0; k0 < CIN; k0 += TK) {
    __syncthreads();
    { // A tile: pure bf16 copy -> async DMA to LDS
      const unsigned short* ga = wq + (size_t)(m0 + ar) * CIN + k0 + ac;
      async_copy_b128(la,      ga);
      async_copy_b128(la + 16, ga + 8);
    }
    { // B tile: x fp32 -> bf16 transposed into Bs[n][k], packed k-pair dword stores
      const float* s0 = x + (((size_t)b * CIN + k0 + k2) * NHW + n0 + nc);
      const float* s1 = s0 + NHW;
      float4 a0 = *(const float4*)(s0), a1 = *(const float4*)(s0 + 4);
      float4 b0 = *(const float4*)(s1), b1 = *(const float4*)(s1 + 4);
      float r0[8] = {a0.x,a0.y,a0.z,a0.w,a1.x,a1.y,a1.z,a1.w};
      float r1[8] = {b0.x,b0.y,b0.z,b0.w,b1.x,b1.y,b1.z,b1.w};
#pragma unroll
      for (int i = 0; i < 8; ++i)
        *(unsigned int*)(Bs + (nc + i) * LDK + k2) = f2bf2(r0[i], r1[i]);
      if (k0 + TK < CIN) __builtin_prefetch(s0 + (size_t)TK * NHW, 0, 3);
    }
    wait_async0();
    __syncthreads();
    v16bf af[2];
    af[0] = frag_a(As + (mt0 * 16 + ln) * LDK, g);
    af[1] = frag_a(As + ((mt0 + 1) * 16 + ln) * LDK, g);
#pragma unroll
    for (int j = 0; j < 4; ++j) {
      v16bf bf = frag_b(Bs + ((nt0 + j) * 16 + ln) * LDK, g);
      acc[0][j] = wmma_bf16(af[0], bf, acc[0][j]);
      acc[1][j] = wmma_bf16(af[1], bf, acc[1][j]);
    }
  }

  const int which = m0 >> 8;                    // 0=q 1=k 2=v (BM=128 divides 256 boundaries)
  unsigned short* dst = which == 0 ? qb : (which == 1 ? kb : vb);
  const int orow0 = m0 & 255;
#pragma unroll
  for (int i = 0; i < 2; ++i)
#pragma unroll
    for (int j = 0; j < 4; ++j) {
      int col = n0 + (nt0 + j) * 16 + ln;
#pragma unroll
      for (int r = 0; r < 8; ++r) {
        int orow = orow0 + (mt0 + i) * 16 + r + g * 8;
        dst[((size_t)b * HID + orow) * NHW + col] = f2bf1(acc[i][j][r]);
      }
    }
}

// ---------------- kernel 3: row softmax on k (fp32 math, bf16 I/O, in place) ----------------
__global__ __launch_bounds__(256) void softmax_rows(unsigned short* __restrict__ kb) {
  __shared__ float red[256];
  const int row = blockIdx.x;
  const int t = threadIdx.x;
  unsigned short* p = kb + (size_t)row * NHW + t * 16;
  const uint4* pu = (const uint4*)p;
  uint4 a = pu[0], b4 = pu[1];
  unsigned int w[8] = {a.x, a.y, a.z, a.w, b4.x, b4.y, b4.z, b4.w};
  float v[16];
#pragma unroll
  for (int i = 0; i < 8; ++i) { v[2*i] = bf2f(w[i]); v[2*i+1] = bf2f(w[i] >> 16); }
  float mx = -1e30f;
#pragma unroll
  for (int i = 0; i < 16; ++i) mx = fmaxf(mx, v[i]);
  red[t] = mx; __syncthreads();
  for (int s = 128; s > 0; s >>= 1) { if (t < s) red[t] = fmaxf(red[t], red[t + s]); __syncthreads(); }
  mx = red[0]; __syncthreads();
  float sum = 0.f;
#pragma unroll
  for (int i = 0; i < 16; ++i) { v[i] = __expf(v[i] - mx); sum += v[i]; }
  red[t] = sum; __syncthreads();
  for (int s = 128; s > 0; s >>= 1) { if (t < s) red[t] += red[t + s]; __syncthreads(); }
  float inv = 1.0f / red[0];
  unsigned int o[8];
#pragma unroll
  for (int i = 0; i < 8; ++i) o[i] = f2bf2(v[2*i] * inv, v[2*i+1] * inv);
  uint4* po = (uint4*)p;
  po[0] = make_uint4(o[0], o[1], o[2], o[3]);
  po[1] = make_uint4(o[4], o[5], o[6], o[7]);
}

// ---------------- kernel 4a: zero context accumulator ----------------
__global__ void zero_f32(float* __restrict__ p, int n) {
  int i = blockIdx.x * 256 + threadIdx.x;
  if (i < n) p[i] = 0.f;
}

// ---------------- kernel 4b: context[d][e] = sum_n ksm[d,n]*v[e,n], split-K + atomics ----------------
__global__ __launch_bounds__(256) void context_kernel(const unsigned short* __restrict__ kb,
                                                      const unsigned short* __restrict__ vb,
                                                      float* __restrict__ ctx) {
  const int split = blockIdx.x, h = blockIdx.y, b = blockIdx.z;
  const int t = threadIdx.x, lane = t & 31, wave = t >> 5;
  const int g = lane >> 4, ln = lane & 15;
  const int mt = wave >> 1, et0 = (wave & 1) * 2;
  v8f acc[2];
  acc[0] = acc[1] = (v8f){0.f,0.f,0.f,0.f,0.f,0.f,0.f,0.f};
  const unsigned short* arow = kb + ((size_t)b * HID + h * DHEAD + mt * 16 + ln) * NHW;
  const unsigned short* brow0 = vb + ((size_t)b * HID + h * DHEAD + et0 * 16 + ln) * NHW;
  const unsigned short* brow1 = brow0 + 16 * NHW;
  for (int nbk = split * NSL; nbk < split * NSL + NSL; nbk += TK) {
    union { v16bf v; uint4 q[2]; } af;
    af.q[0] = *(const uint4*)(arow + nbk + (g << 3));
    af.q[1] = *(const uint4*)(arow + nbk + 16 + (g << 3));
    union { v16bf v; uint4 q[2]; } bf0, bf1;
    bf0.q[0] = *(const uint4*)(brow0 + nbk + (g << 3));
    bf0.q[1] = *(const uint4*)(brow0 + nbk + 16 + (g << 3));
    bf1.q[0] = *(const uint4*)(brow1 + nbk + (g << 3));
    bf1.q[1] = *(const uint4*)(brow1 + nbk + 16 + (g << 3));
    acc[0] = wmma_bf16(af.v, bf0.v, acc[0]);
    acc[1] = wmma_bf16(af.v, bf1.v, acc[1]);
  }
#pragma unroll
  for (int j = 0; j < 2; ++j) {
    int e = (et0 + j) * 16 + ln;
#pragma unroll
    for (int r = 0; r < 8; ++r) {
      int d = mt * 16 + r + g * 8;
      atomicAdd(&ctx[(((size_t)b * HEADS + h) * DHEAD + d) * DHEAD + e], acc[j][r]);
    }
  }
}

// ---------------- kernel 5: W_eff[o][h*64+d] = sum_e w_out[o][h*64+e] * ctx[b][h][d][e] ----------------
__global__ __launch_bounds__(256) void weff_kernel(const float* __restrict__ wout,
                                                   const float* __restrict__ ctx,
                                                   unsigned short* __restrict__ weff) {
  int idx = blockIdx.x * 256 + threadIdx.x;     // < 16*256*256
  int b = idx >> 16;
  int o = (idx >> 8) & 255;
  int c = idx & 255;
  int h = c >> 6, d = c & 63;
  const float* wr = wout + (size_t)o * HID + h * DHEAD;
  const float* cr = ctx + (((size_t)b * HEADS + h) * DHEAD + d) * DHEAD;
  float acc = 0.f;
#pragma unroll 8
  for (int e = 0; e < DHEAD; ++e) acc = fmaf(wr[e], cr[e], acc);
  weff[idx] = f2bf1(acc);
}

// ---------------- kernel 6: y = gamma*(W_eff @ q + b_out) + x ----------------
__global__ __launch_bounds__(256) void out_gemm(const unsigned short* __restrict__ weff,
                                                const unsigned short* __restrict__ qb,
                                                const float* __restrict__ bout,
                                                const float* __restrict__ gamma,
                                                const float* __restrict__ x,
                                                float* __restrict__ y) {
  __shared__ alignas(16) unsigned short As[BM * LDK];
  __shared__ alignas(16) unsigned short Bs[BN * LDK];
  const int n0 = blockIdx.x * BN;
  const int m0 = blockIdx.y * BM;
  const int b  = blockIdx.z;
  const int t  = threadIdx.x;
  const int lane = t & 31, wave = t >> 5;
  const int g = lane >> 4, ln = lane & 15;
  const int mt0 = (wave >> 1) * 2, nt0 = (wave & 1) * 4;
  const unsigned short* wA = weff + (size_t)b * HID * HID;

  v8f acc[2][4];
#pragma unroll
  for (int i = 0; i < 2; ++i)
#pragma unroll
    for (int j = 0; j < 4; ++j) acc[i][j] = (v8f){0.f,0.f,0.f,0.f,0.f,0.f,0.f,0.f};

  const int ar = t >> 1, ac = (t & 1) * 16;
  const int k2 = (t >> 4) * 2, nc = (t & 15) * 8;
  const unsigned la = (unsigned)(uintptr_t)(As + ar * LDK + ac);

  for (int k0 = 0; k0 < HID; k0 += TK) {
    __syncthreads();
    { // A tile (bf16): async DMA to LDS
      const unsigned short* ga = wA + (size_t)(m0 + ar) * HID + k0 + ac;
      async_copy_b128(la,      ga);
      async_copy_b128(la + 16, ga + 8);
    }
    { // B tile from bf16 q: load 2 rows x 8 cols, repack transposed k-pair dwords
      const unsigned int* s0 = (const unsigned int*)(qb + (((size_t)b * HID + k0 + k2) * NHW + n0 + nc));
      const unsigned int* s1 = s0 + (NHW >> 1);
      uint4 ra = *(const uint4*)s0;
      uint4 rb = *(const uint4*)s1;
      unsigned int r0[4] = {ra.x, ra.y, ra.z, ra.w};
      unsigned int r1[4] = {rb.x, rb.y, rb.z, rb.w};
#pragma unroll
      for (int jj = 0; jj < 4; ++jj) {
        *(unsigned int*)(Bs + (nc + 2*jj)     * LDK + k2) = (r0[jj] & 0xFFFFu) | (r1[jj] << 16);
        *(unsigned int*)(Bs + (nc + 2*jj + 1) * LDK + k2) = (r0[jj] >> 16) | (r1[jj] & 0xFFFF0000u);
      }
      if (k0 + TK < HID) __builtin_prefetch(s0 + (size_t)TK * (NHW >> 1), 0, 3);
    }
    wait_async0();
    __syncthreads();
    v16bf af[2];
    af[0] = frag_a(As + (mt0 * 16 + ln) * LDK, g);
    af[1] = frag_a(As + ((mt0 + 1) * 16 + ln) * LDK, g);
#pragma unroll
    for (int j = 0; j < 4; ++j) {
      v16bf bf = frag_b(Bs + ((nt0 + j) * 16 + ln) * LDK, g);
      acc[0][j] = wmma_bf16(af[0], bf, acc[0][j]);
      acc[1][j] = wmma_bf16(af[1], bf, acc[1][j]);
    }
  }

#pragma unroll
  for (int i = 0; i < 2; ++i)
#pragma unroll
    for (int j = 0; j < 4; ++j) {
      int col = n0 + (nt0 + j) * 16 + ln;
#pragma unroll
      for (int r = 0; r < 8; ++r) {
        int o = m0 + (mt0 + i) * 16 + r + g * 8;
        size_t off = ((size_t)b * HID + o) * NHW + col;
        y[off] = (acc[i][j][r] + bout[o]) * gamma[o] + x[off];
      }
    }
}

// ---------------- launch ----------------
extern "C" void kernel_launch(void* const* d_in, const int* in_sizes, int n_in,
                              void* d_out, int out_size, void* d_ws, size_t ws_size,
                              hipStream_t stream) {
  const float* x     = (const float*)d_in[0];
  const float* w_qkv = (const float*)d_in[1];
  const float* w_out = (const float*)d_in[2];
  const float* b_out = (const float*)d_in[3];
  const float* gamma = (const float*)d_in[4];
  float* y = (float*)d_out;

  char* ws = (char*)d_ws;
  const size_t QB   = (size_t)NB * HID * NHW * sizeof(unsigned short); // 32 MiB each
  const size_t CTXB = (size_t)NB * HEADS * DHEAD * DHEAD * sizeof(float);
  const size_t WEB  = (size_t)NB * HID * HID * sizeof(unsigned short);
  unsigned short* qb   = (unsigned short*)(ws);
  unsigned short* kb   = (unsigned short*)(ws + QB);
  unsigned short* vb   = (unsigned short*)(ws + 2 * QB);
  float*          ctx  = (float*)(ws + 3 * QB);
  unsigned short* weff = (unsigned short*)(ws + 3 * QB + CTXB);
  unsigned short* wqb  = (unsigned short*)(ws + 3 * QB + CTXB + WEB);

  cvt_to_bf16<<<(OQKV * CIN + 255) / 256, 256, 0, stream>>>(w_qkv, wqb, OQKV * CIN);
  qkv_gemm<<<dim3(NHW / BN, OQKV / BM, NB), 256, 0, stream>>>(wqb, x, qb, kb, vb);
  softmax_rows<<<NB * HID, 256, 0, stream>>>(kb);
  zero_f32<<<(NB * HEADS * DHEAD * DHEAD + 255) / 256, 256, 0, stream>>>(ctx, NB * HEADS * DHEAD * DHEAD);
  context_kernel<<<dim3(NSPLIT, HEADS, NB), 256, 0, stream>>>(kb, vb, ctx);
  weff_kernel<<<NB * HID * HID / 256, 256, 0, stream>>>(w_out, ctx, weff);
  out_gemm<<<dim3(NHW / BN, HID / BM, NB), 256, 0, stream>>>(weff, qb, b_out, gamma, x, y);
}
